// L0Adapter_3564822856005
// MI455X (gfx1250) — matrix operand order repaired and checked by reading, not compiled, not run on hardware
//
#include <hip/hip_runtime.h>

// ---------------------------------------------------------------------------
// Fused L0-gated adapter for MI455X (gfx1250, wave32, WMMA).
//   out = x + ( relu( LN(x)*dmask @ down_w^T + down_b ) * umask ) @ up_w^T + up_b
// bf16 WMMA (v_wmma_f32_16x16x32_bf16), f32 accumulation, LDS-resident
// intermediate. One workgroup = 64 tokens, 256 threads (8 waves).
// ---------------------------------------------------------------------------

typedef __bf16 bf16_t;
typedef __attribute__((ext_vector_type(16))) __bf16 v16bf;
typedef __attribute__((ext_vector_type(8)))  float  v8f;

union V16U { v16bf v; uint4 q[2]; };

#define D_IN    2048
#define BN      512
#define M_TILE  64
#define KC      64
#define N_TOK   16384   // 8 * 2048

__global__ __launch_bounds__(256)
void adapter_fused(const float* __restrict__ x,
                   const float* __restrict__ down_w,   // [BN, D_IN]
                   const float* __restrict__ down_b,   // [BN]
                   const float* __restrict__ up_w,     // [D_IN, BN]
                   const float* __restrict__ up_b,     // [D_IN]
                   const float* __restrict__ ln_g,     // [D_IN]
                   const float* __restrict__ ln_b,     // [D_IN]
                   const float* __restrict__ down_loga,// [16, D_IN]
                   const float* __restrict__ up_loga,  // [16, BN]
                   const int*   __restrict__ lang_id,  // [1]
                   float* __restrict__ out)            // [N_TOK, D_IN]
{
    __shared__ __align__(32) bf16_t A_s[M_TILE * KC];     //  8 KB: LN'd act chunk
    __shared__ __align__(32) bf16_t B_s[BN * KC];         // 64 KB: down_w chunk
    __shared__ __align__(32) bf16_t H_s[M_TILE * BN];     // 64 KB: bottleneck acts
    __shared__ float ascale_s[D_IN];                      //  8 KB: ln_g*dmask
    __shared__ float abias_s[D_IN];                       //  8 KB: ln_b*dmask
    __shared__ float umask_s[BN];
    __shared__ float dbias_s[BN];
    __shared__ float mu_s[M_TILE];
    __shared__ float rs_s[M_TILE];
    __shared__ float red1_s[256];
    __shared__ float red2_s[256];

    const int tid   = threadIdx.x;
    const int wave  = tid >> 5;
    const int lane  = tid & 31;
    const int l16   = lane & 15;
    const int lhalf = lane >> 4;          // 0: lanes 0-15, 1: lanes 16-31
    const int row0  = blockIdx.x * M_TILE;
    const int lid   = lang_id[0];

    // ---------------- Phase 0a: per-feature gates -> fused LN coefficients
    #pragma unroll
    for (int i = 0; i < 8; ++i) {
        int d = tid + i * 256;
        float z = down_loga[lid * D_IN + d];
        float s = 1.f / (1.f + __expf(-z));
        float m = fminf(fmaxf(s * 1.2f - 0.1f, 0.f), 1.f);   // zeta-gamma=1.2
        ascale_s[d] = ln_g[d] * m;
        abias_s[d]  = ln_b[d] * m;
    }
    #pragma unroll
    for (int i = 0; i < 2; ++i) {
        int k = tid + i * 256;
        float z = up_loga[lid * BN + k];
        float s = 1.f / (1.f + __expf(-z));
        umask_s[k] = fminf(fmaxf(s * 1.2f - 0.1f, 0.f), 1.f);
        dbias_s[k] = down_b[k];
    }
    // ---------------- Phase 0b: LayerNorm statistics (4 threads per row)
    {
        int r = tid >> 2, q = tid & 3;
        const float4* xp = (const float4*)(x + (size_t)(row0 + r) * D_IN + q * 512);
        float s = 0.f, s2 = 0.f;
        for (int j = 0; j < 128; ++j) {
            float4 f = xp[j];
            s  += f.x + f.y + f.z + f.w;
            s2 += f.x * f.x + f.y * f.y + f.z * f.z + f.w * f.w;
        }
        red1_s[tid] = s; red2_s[tid] = s2;
    }
    __syncthreads();
    if (tid < M_TILE) {
        float s  = red1_s[tid*4] + red1_s[tid*4+1] + red1_s[tid*4+2] + red1_s[tid*4+3];
        float s2 = red2_s[tid*4] + red2_s[tid*4+1] + red2_s[tid*4+2] + red2_s[tid*4+3];
        float mu  = s * (1.f / D_IN);
        float var = s2 * (1.f / D_IN) - mu * mu;
        mu_s[tid] = mu;
        rs_s[tid] = rsqrtf(var + 1e-5f);
    }
    __syncthreads();

    // ---------------- GEMM1: [64 x 2048] @ [2048 x 512], K-chunked.
    // Each wave: all 4 M-tiles x its own 64 N-columns (no weight redundancy).
    v8f acc1[16];
    #pragma unroll
    for (int t = 0; t < 16; ++t)
        #pragma unroll
        for (int e = 0; e < 8; ++e) acc1[t][e] = 0.f;

    #pragma unroll 1
    for (int kc = 0; kc < D_IN / KC; ++kc) {
        const int kb = kc * KC;
        {   // stage A chunk: normalize + gate + cvt bf16
            int r = tid >> 2, c0 = (tid & 3) << 4;
            const float4* xp = (const float4*)(x + (size_t)(row0 + r) * D_IN + kb + c0);
            float mu = mu_s[r], rs = rs_s[r];
            #pragma unroll
            for (int j = 0; j < 4; ++j) {
                float4 f = xp[j];
                int c = c0 + j * 4;
                A_s[r*KC + c + 0] = (bf16_t)((f.x - mu)*rs*ascale_s[kb+c+0] + abias_s[kb+c+0]);
                A_s[r*KC + c + 1] = (bf16_t)((f.y - mu)*rs*ascale_s[kb+c+1] + abias_s[kb+c+1]);
                A_s[r*KC + c + 2] = (bf16_t)((f.z - mu)*rs*ascale_s[kb+c+2] + abias_s[kb+c+2]);
                A_s[r*KC + c + 3] = (bf16_t)((f.w - mu)*rs*ascale_s[kb+c+3] + abias_s[kb+c+3]);
            }
        }
        {   // stage down_w chunk: fp32 -> bf16 (each weight read once per WG)
            #pragma unroll
            for (int rep = 0; rep < 2; ++rep) {
                int n = tid + rep * 256;
                const float4* wp = (const float4*)(down_w + (size_t)n * D_IN + kb);
                #pragma unroll
                for (int j = 0; j < 16; ++j) {
                    float4 f = wp[j];
                    B_s[n*KC + j*4 + 0] = (bf16_t)f.x;
                    B_s[n*KC + j*4 + 1] = (bf16_t)f.y;
                    B_s[n*KC + j*4 + 2] = (bf16_t)f.z;
                    B_s[n*KC + j*4 + 3] = (bf16_t)f.w;
                }
            }
        }
        __syncthreads();
        #pragma unroll
        for (int ks = 0; ks < 2; ++ks) {           // two k=32 steps per chunk
            V16U af[4];
            #pragma unroll
            for (int mt = 0; mt < 4; ++mt) {       // A frag: 16 contiguous K
                int base = (mt * 16 + l16) * KC + ks * 32 + lhalf * 16;
                af[mt].q[0] = *(const uint4*)(A_s + base);
                af[mt].q[1] = *(const uint4*)(A_s + base + 8);
            }
            #pragma unroll
            for (int nt = 0; nt < 4; ++nt) {
                int n  = wave * 64 + nt * 16 + l16;
                int k0 = ks * 32 + lhalf * 16;     // B frag: 16 contiguous K
                v16bf bfrag = *(const v16bf*)(B_s + n * KC + k0);
                #pragma unroll
                for (int mt = 0; mt < 4; ++mt)
                    acc1[nt*4 + mt] = __builtin_amdgcn_wmma_f32_16x16x32_bf16(
                        false, af[mt].v, false, bfrag, (short)0,
                        acc1[nt*4 + mt], false, false);
            }
        }
        __syncthreads();
    }

    // ---------------- Epilogue 1: bias + ReLU + umask -> bf16 H tile in LDS
    #pragma unroll
    for (int nt = 0; nt < 4; ++nt) {
        int n = wave * 64 + nt * 16 + l16;
        float um = umask_s[n], db = dbias_s[n];
        #pragma unroll
        for (int mt = 0; mt < 4; ++mt) {
            v8f a = acc1[nt*4 + mt];
            #pragma unroll
            for (int v = 0; v < 8; ++v) {
                int r = mt * 16 + lhalf * 8 + v;   // C layout: M = v + 8*half
                float h = a[v] + db;
                h = h > 0.f ? h : 0.f;
                H_s[r * BN + n] = (bf16_t)(h * um);
            }
        }
    }
    __syncthreads();

    // ---------------- GEMM2: [64 x 512] @ [512 x 2048] + bias + residual.
    // Each wave: all 4 M-tiles x its own 256 output columns.
    #pragma unroll 1
    for (int nt2 = 0; nt2 < 16; ++nt2) {
        const int n0 = wave * 256 + nt2 * 16;
        const int d  = n0 + l16;
        const float* wp = up_w + (size_t)d * BN;
        if (nt2 + 1 < 16)                          // uniform guard
            __builtin_prefetch(up_w + (size_t)(d + 16) * BN, 0, 1);
        v8f acc2[4];
        #pragma unroll
        for (int mt = 0; mt < 4; ++mt)
            #pragma unroll
            for (int e = 0; e < 8; ++e) acc2[mt][e] = 0.f;

        #pragma unroll 2
        for (int ks = 0; ks < 16; ++ks) {
            int k0 = ks * 32 + lhalf * 16;
            v16bf bfrag;                           // 16 contiguous fp32 -> bf16
            #pragma unroll
            for (int j = 0; j < 4; ++j) {
                float4 f = *(const float4*)(wp + k0 + j * 4);
                bfrag[j*4 + 0] = (__bf16)f.x;
                bfrag[j*4 + 1] = (__bf16)f.y;
                bfrag[j*4 + 2] = (__bf16)f.z;
                bfrag[j*4 + 3] = (__bf16)f.w;
            }
            #pragma unroll
            for (int mt = 0; mt < 4; ++mt) {
                int base = (mt * 16 + l16) * BN + ks * 32 + lhalf * 16;
                V16U af;
                af.q[0] = *(const uint4*)(H_s + base);
                af.q[1] = *(const uint4*)(H_s + base + 8);
                acc2[mt] = __builtin_amdgcn_wmma_f32_16x16x32_bf16(
                    false, af.v, false, bfrag, (short)0, acc2[mt], false, false);
            }
        }
        float ub = up_b[d];
        #pragma unroll
        for (int mt = 0; mt < 4; ++mt) {
            #pragma unroll
            for (int v = 0; v < 8; ++v) {
                int r = row0 + mt * 16 + lhalf * 8 + v;
                size_t idx = (size_t)r * D_IN + d;
                out[idx] = acc2[mt][v] + ub + x[idx];
            }
        }
    }
}

extern "C" void kernel_launch(void* const* d_in, const int* in_sizes, int n_in,
                              void* d_out, int out_size, void* d_ws, size_t ws_size,
                              hipStream_t stream) {
    const float* x         = (const float*)d_in[0];
    const float* down_w    = (const float*)d_in[1];
    const float* down_b    = (const float*)d_in[2];
    const float* up_w      = (const float*)d_in[3];
    const float* up_b      = (const float*)d_in[4];
    const float* ln_g      = (const float*)d_in[5];
    const float* ln_b      = (const float*)d_in[6];
    const float* down_loga = (const float*)d_in[7];
    const float* up_loga   = (const float*)d_in[8];
    const int*   lang_id   = (const int*)d_in[9];
    float* out = (float*)d_out;

    dim3 grid(N_TOK / M_TILE);   // 256 workgroups
    dim3 block(256);             // 8 wave32 waves
    adapter_fused<<<grid, block, 0, stream>>>(x, down_w, down_b, up_w, up_b,
                                              ln_g, ln_b, down_loga, up_loga,
                                              lang_id, out);
}
